// SelfAttentionHead_5033701671366
// MI455X (gfx1250) — matrix operand order repaired and checked by reading, not compiled, not run on hardware
//
#include <hip/hip_runtime.h>

typedef _Float16 f16;
typedef __attribute__((ext_vector_type(16))) _Float16 v16h;
typedef __attribute__((ext_vector_type(8)))  _Float16 v8h;
typedef __attribute__((ext_vector_type(8)))  float    v8f;
typedef __attribute__((ext_vector_type(4)))  float    v4f;

#define NB 8
#define NT 2048
#define NC 1024
#define NH 64
#define BT (NB*NT)

// ---------------- workspace layout (f16 elements) ----------------
// [0)              Wt : 3 * NH * NC   (K,Q,V weights, transposed [H][C])
// [WT_SZ)          Kh : BT * NH       (K, row major [token][h])
// [.. )            Qh : BT * NH       (Q, row major)
// [.. )            Vt : NB * NH * NT  (V, transposed [b][h][t])
#define WT_SZ ((size_t)3 * NH * NC)
#define KQ_SZ ((size_t)BT * NH)

static __device__ inline v16h pack16(v8h lo, v8h hi) {
  v16h r;
#pragma unroll
  for (int i = 0; i < 8; ++i) { r[i] = lo[i]; r[8 + i] = hi[i]; }
  return r;
}

// A operand (16-bit 16x32): lane row = l&15 ; K = {b..b+7, 16+b..16+b+7}, b = 8*(l>>4)
static __device__ inline v16h load_matA_f16(const f16* p, int half) {
  const int b = half * 8;
  v8h lo = *(const v8h*)(p + b);
  v8h hi = *(const v8h*)(p + 16 + b);
  return pack16(lo, hi);
}

// B operand (16-bit 32x16): lane col = l&15 ; K = {b..b+7, 16+b..16+b+7}, b = 8*(l>>4)
static __device__ inline v16h load_matB_f16(const f16* p, int half) {
  const f16* q = p + half * 8;
  v8h lo = *(const v8h*)(q);
  v8h hi = *(const v8h*)(q + 16);
  return pack16(lo, hi);
}

// A operand built from fp32 source (convert on the fly)
static __device__ inline v16h load_matA_f32(const float* p, int half) {
  const int b = half * 8;
  v4f a0 = *(const v4f*)(p + b);
  v4f a1 = *(const v4f*)(p + b + 4);
  v4f a2 = *(const v4f*)(p + 16 + b);
  v4f a3 = *(const v4f*)(p + 16 + b + 4);
  v16h r;
#pragma unroll
  for (int i = 0; i < 4; ++i) {
    r[i]      = (f16)a0[i];
    r[4 + i]  = (f16)a1[i];
    r[8 + i]  = (f16)a2[i];
    r[12 + i] = (f16)a3[i];
  }
  return r;
}

// ---------------- stage 0: weights fp32 [C,H] -> f16 transposed [H,C] ----------------
__global__ void cvt_w_kernel(const float* __restrict__ Wk,
                             const float* __restrict__ Wq,
                             const float* __restrict__ Wv,
                             f16* __restrict__ wt) {
  int i = blockIdx.x * blockDim.x + threadIdx.x;   // 0 .. 3*NH*NC-1
  int mat = i / (NH * NC);
  int rem = i - mat * (NH * NC);
  int h = rem / NC;
  int c = rem - h * NC;
  const float* W = (mat == 0) ? Wk : ((mat == 1) ? Wq : Wv);
  wt[i] = (f16)W[c * NH + h];
}

// ---------------- stage 1: fused K/Q/V projection ----------------
// one wave per 16-token tile; 12 WMMAs (3 mats x 4 h-tiles) per 32-wide k step.
__global__ __launch_bounds__(256) void proj16_kernel(const float* __restrict__ x,
                                                     const f16* __restrict__ wt,
                                                     f16* __restrict__ kh,
                                                     f16* __restrict__ qh,
                                                     f16* __restrict__ vt) {
  const int tid  = threadIdx.x;
  const int lane = tid & 31;
  const int wave = tid >> 5;
  const int wid  = blockIdx.x * (blockDim.x >> 5) + wave;   // 0..1023
  const int row  = lane & 15;
  const int half = lane >> 4;

  const long rowg = (long)wid * 16 + row;                   // token row for A operand
  const float* xr = x + rowg * NC;

  v8f acc[3][4] = {};
  for (int kc = 0; kc < NC; kc += 32) {
    v16h a = load_matA_f32(xr + kc, half);
#pragma unroll
    for (int m = 0; m < 3; ++m) {
      const f16* wm = wt + (size_t)m * NH * NC;
#pragma unroll
      for (int n = 0; n < 4; ++n) {
        // B tile: K-dim = c (contiguous in Wt rows), col = h = n*16 + (lane&15)
        const f16* wrow = wm + (size_t)(n * 16 + row) * NC + kc;
        v16h b = load_matB_f16(wrow, half);
        acc[m][n] = __builtin_amdgcn_wmma_f32_16x16x32_f16(
            false, a, false, b, (short)0, acc[m][n], false, false);
      }
    }
  }

  // D layout: lane col = lane&15 ; VGPR r -> row r + 8*half
#pragma unroll
  for (int n = 0; n < 4; ++n) {
    const int hcol = n * 16 + row;
#pragma unroll
    for (int r = 0; r < 8; ++r) {
      const long trow = (long)wid * 16 + r + 8 * half;
      kh[trow * NH + hcol] = (f16)acc[0][n][r];
      qh[trow * NH + hcol] = (f16)acc[1][n][r];
      const int bb = (int)(trow / NT);
      const int tt = (int)(trow - (long)bb * NT);
      vt[((size_t)bb * NH + hcol) * NT + tt] = (f16)acc[2][n][r];
    }
  }
}

// ---------------- stage 2: fused causal flash attention ----------------
// one wave per (batch, 16-query tile); 32-key blocks; online softmax.
__global__ __launch_bounds__(256) void attn16_kernel(const f16* __restrict__ qh,
                                                     const f16* __restrict__ kh,
                                                     const f16* __restrict__ vt,
                                                     float* __restrict__ out) {
  __shared__ f16 ldsP[8 * 16 * 40];           // per-wave 16x32 P tile, row stride 40
  const int tid  = threadIdx.x;
  const int lane = tid & 31;
  const int wave = tid >> 5;
  const int wid  = blockIdx.x * 8 + wave;     // 0..1023
  const int b    = wid >> 7;                  // / (NT/16)
  const int qt   = wid & 127;
  const int q0   = qt * 16;
  const int row  = lane & 15;
  const int half = lane >> 4;
  f16* pl = ldsP + wave * (16 * 40);

  // preload Q A-operands for the two 32-wide h steps (kept in registers)
  const f16* qrow = qh + ((size_t)b * NT + q0 + row) * NH;
  v16h aq0 = load_matA_f16(qrow, half);
  v16h aq1 = load_matA_f16(qrow + 32, half);

  float m_i[8], l_i[8];
  v8f o[4] = {};
#pragma unroll
  for (int r = 0; r < 8; ++r) { m_i[r] = -3.0e38f; l_i[r] = 0.0f; }

  const float scale = 0.03125f;               // C^-0.5 = 1/sqrt(1024)
  const int nkeys = q0 + 16;                  // causal: keys 0 .. q0+15

  for (int kb = 0; kb < nkeys; kb += 32) {
    float s[2][8];
#pragma unroll
    for (int st = 0; st < 2; ++st) {
      const int k0 = kb + st * 16;
      if (k0 < nkeys) {                       // wave-uniform branch
        const f16* krow = kh + ((size_t)b * NT + k0 + row) * NH;  // col = key = lane&15
        v8f sacc = {};
        v16h bk0 = load_matB_f16(krow, half);
        sacc = __builtin_amdgcn_wmma_f32_16x16x32_f16(
            false, aq0, false, bk0, (short)0, sacc, false, false);
        v16h bk1 = load_matB_f16(krow + 32, half);
        sacc = __builtin_amdgcn_wmma_f32_16x16x32_f16(
            false, aq1, false, bk1, (short)0, sacc, false, false);
#pragma unroll
        for (int r = 0; r < 8; ++r) {
          const int qg = q0 + r + 8 * half;
          const int kg = k0 + row;
          s[st][r] = (kg <= qg) ? sacc[r] * scale : -3.0e38f;
        }
      } else {
#pragma unroll
        for (int r = 0; r < 8; ++r) s[st][r] = -3.0e38f;
      }
    }

    // online softmax; rows of a C/D VGPR live across one 16-lane group
    float corr[8];
#pragma unroll
    for (int r = 0; r < 8; ++r) {
      float mx = fmaxf(s[0][r], s[1][r]);
      mx = fmaxf(mx, __shfl_xor(mx, 1));
      mx = fmaxf(mx, __shfl_xor(mx, 2));
      mx = fmaxf(mx, __shfl_xor(mx, 4));
      mx = fmaxf(mx, __shfl_xor(mx, 8));
      const float mnew = fmaxf(m_i[r], mx);
      const float p0 = __expf(s[0][r] - mnew);
      const float p1 = __expf(s[1][r] - mnew);
      corr[r] = __expf(m_i[r] - mnew);
      m_i[r] = mnew;
      float ps = p0 + p1;
      ps += __shfl_xor(ps, 1);
      ps += __shfl_xor(ps, 2);
      ps += __shfl_xor(ps, 4);
      ps += __shfl_xor(ps, 8);
      l_i[r] = l_i[r] * corr[r] + ps;
      // C/D layout -> LDS (row m = r+8*half, cols = st*16 + lane&15)
      pl[(r + 8 * half) * 40 + row]      = (f16)p0;
      pl[(r + 8 * half) * 40 + 16 + row] = (f16)p1;
    }

#pragma unroll
    for (int n = 0; n < 4; ++n)
#pragma unroll
      for (int r = 0; r < 8; ++r) o[n][r] *= corr[r];

    // reload P in A layout (intra-wave LDS ops stay in order)
    v16h ap = load_matA_f16(pl + row * 40, half);

    // PV: B tile from transposed V (K-dim = key, contiguous in Vt rows)
#pragma unroll
    for (int n = 0; n < 4; ++n) {
      const f16* vrow = vt + ((size_t)b * NH + n * 16 + row) * NT + kb;
      v16h bv = load_matB_f16(vrow, half);
      o[n] = __builtin_amdgcn_wmma_f32_16x16x32_f16(
          false, ap, false, bv, (short)0, o[n], false, false);
    }
  }

  // epilogue: normalize and store fp32 output
#pragma unroll
  for (int n = 0; n < 4; ++n) {
    const int hcol = n * 16 + row;
#pragma unroll
    for (int r = 0; r < 8; ++r) {
      const float inv = 1.0f / l_i[r];
      out[((size_t)b * NT + q0 + r + 8 * half) * NH + hcol] = o[n][r] * inv;
    }
  }
}

extern "C" void kernel_launch(void* const* d_in, const int* in_sizes, int n_in,
                              void* d_out, int out_size, void* d_ws, size_t ws_size,
                              hipStream_t stream) {
  (void)in_sizes; (void)n_in; (void)out_size; (void)ws_size;
  const float* x  = (const float*)d_in[0];
  const float* Wk = (const float*)d_in[1];
  const float* Wq = (const float*)d_in[2];
  const float* Wv = (const float*)d_in[3];
  float* out = (float*)d_out;

  f16* ws = (f16*)d_ws;
  f16* wt = ws;                          // 3*NH*NC
  f16* kh = ws + WT_SZ;                  // BT*NH
  f16* qh = kh + KQ_SZ;                  // BT*NH
  f16* vt = qh + KQ_SZ;                  // NB*NH*NT

  // stage 0: weight convert/transpose (3*NH*NC elements)
  cvt_w_kernel<<<(3 * NH * NC) / 256, 256, 0, stream>>>(Wk, Wq, Wv, wt);

  // stage 1: fused QKV projection — 1024 waves (BT/16), 8 waves/block
  proj16_kernel<<<(BT / 16) / 8, 256, 0, stream>>>(x, wt, kh, qh, vt);

  // stage 2: flash attention — 1024 waves (NB * NT/16), 8 waves/block
  attn16_kernel<<<(NB * (NT / 16)) / 8, 256, 0, stream>>>(qh, kh, vt, out);
}
